// MoChA_24893630448015
// MI455X (gfx1250) — compile-verified, matrix operand order checked by reading, not run on hardware
//
#include <hip/hip_runtime.h>
#include <math.h>

// ---------------- problem constants (fixed by reference) ----------------
#define B_    8
#define QLEN  128
#define KLEN  1024
#define D_    512
#define HM_   4
#define HC_   4
#define DK_   128          // D_/HM_
#define HTOT  16           // HM_*HC_
#define VD    32           // D_/HTOT
#define CHUNK 4
#define EPS_  1e-6f
#define NEG_INF_ (-3.402823466e+38f)

typedef __bf16 bf16;
typedef __attribute__((ext_vector_type(16))) __bf16 v16bf;
typedef __attribute__((ext_vector_type(8)))  float  v8f;

// ---------------- WMMA operand load (ISA 7.12.2 16-bit A/B layout) ------
// tile points at element (0,0) of a 16x32 slab; ld in elements.
// lane L: row/col = L%16 ; elements 0..7 -> k = (L/16)*8 + 0..7,
//         elements 8..15 -> k = 16 + (L/16)*8 + 0..7  => two b128 loads.
// Works for global or LDS-backed pointers (compiler picks vmem vs ds op).
__device__ inline v16bf load_ab16x32(const bf16* __restrict__ tile, int ld) {
  const int lane = threadIdx.x & 31;
  const int r    = lane & 15;
  const int half = lane >> 4;
  const bf16* p = tile + (size_t)r * ld + half * 8;
  union { uint4 u[2]; v16bf v; } t;
  t.u[0] = *reinterpret_cast<const uint4*>(p);
  t.u[1] = *reinterpret_cast<const uint4*>(p + 16);
  return t.v;
}

__device__ inline v8f wmma_bf16(v16bf a, v16bf b, v8f acc) {
  return __builtin_amdgcn_wmma_f32_16x16x32_bf16(
      /*neg_a=*/false, a, /*neg_b=*/false, b,
      /*c_mod=*/(short)0, acc, /*reuse_a=*/false, /*reuse_b=*/false);
}

// ---------------- CDNA5 async Global->LDS staging -----------------------
// ISA 15.18.3 opcode 98: per-lane 16B copy memory->LDS, tracked by ASYNCcnt.
// lds_off = wave-relative LDS byte offset (low 32 bits of generic pointer).
__device__ inline void async_copy_b128(unsigned lds_off, const void* gptr) {
  asm volatile("global_load_async_to_lds_b128 %0, %1, off"
               :: "v"(lds_off), "v"((unsigned long long)(size_t)gptr)
               : "memory");
}
__device__ inline void wait_async0() {
#if __has_builtin(__builtin_amdgcn_s_wait_asynccnt)
  __builtin_amdgcn_s_wait_asynccnt(0);
#else
  asm volatile("s_wait_asynccnt 0x0" ::: "memory");
#endif
}

// ---------------- pack / transpose-pack to bf16 -------------------------
__global__ void pack_bf16_kernel(const float* __restrict__ src,
                                 bf16* __restrict__ dst, int n) {
  int i = blockIdx.x * blockDim.x + threadIdx.x;
  int stride = gridDim.x * blockDim.x;
  for (; i < n; i += stride) dst[i] = (bf16)src[i];
}

// W[K,N] f32 -> Wt[n*K + k] bf16 (N-major so B-operand reads are contiguous)
__global__ void packT_bf16_kernel(const float* __restrict__ W,
                                  bf16* __restrict__ Wt, int K, int N) {
  int i = blockIdx.x * blockDim.x + threadIdx.x;
  if (i < K * N) {
    int k = i / N, n = i % N;
    Wt[(size_t)n * K + k] = (bf16)W[i];
  }
}

// ---------------- generic WMMA GEMM (Kd fixed = 512) --------------------
// C[m,n] = sum_k A[m,k]*Bt[n,k] + bias. blockDim (32,8): 8 waves, each a
// 32x32 output tile (2x2 WMMA). All 8 waves share one 32x512 B panel ->
// async-staged into LDS once per block; A double-buffered from global.
// mode 0: store bf16 row-major Cb[m*ldc+n]
// mode 1: v-projection transpose store: Cb[(b*512 + n)*1024 + k], m=b*1024+k
// mode 2: store f32 Cf[m*ldc+n]
__global__ void gemm_wmma_kernel(const bf16* __restrict__ A, int lda,
                                 const bf16* __restrict__ Bt, int ldb,
                                 const float* __restrict__ bias,
                                 float* __restrict__ Cf,
                                 bf16* __restrict__ Cb, int ldc,
                                 int M, int mode) {
  __shared__ __align__(16) bf16 Bs[32 * D_];          // 32 KB
  const int lane = threadIdx.x;
  const int n0 = blockIdx.x * 32;
  const int t  = threadIdx.y * 32 + lane;

  // ---- async-stage B panel: 32 rows x 1024B, 2048 x 16B chunks ----
  {
    const char* Bg = (const char*)(Bt + (size_t)n0 * ldb);
    const unsigned lds_base = (unsigned)(size_t)(const void*)&Bs[0];
    #pragma unroll
    for (int it = 0; it < 8; ++it) {
      int idx = it * 256 + t;
      int rn = idx >> 6;              // row in panel
      int cb = idx & 63;              // 16B chunk in row
      async_copy_b128(lds_base + (unsigned)(rn * (D_ * 2) + cb * 16),
                      Bg + (size_t)rn * ldb * 2 + cb * 16);
    }
  }
  wait_async0();
  __syncthreads();

  const int m0 = (blockIdx.y * 8 + threadIdx.y) * 32;
  if (m0 >= M) return;

  const bf16* A0 = A  + (size_t)m0 * lda;
  const bf16* A1 = A0 + (size_t)16 * lda;

  v8f acc00 = {}, acc01 = {}, acc10 = {}, acc11 = {};
  v16bf a0 = load_ab16x32(A0, lda);
  v16bf a1 = load_ab16x32(A1, lda);

  for (int kk = 0; kk < D_; kk += 32) {
    const int kn = (kk + 32 < D_) ? kk + 32 : kk;   // next A slab (dup on last)
    v16bf na0 = load_ab16x32(A0 + kn, lda);
    v16bf na1 = load_ab16x32(A1 + kn, lda);
    if (kk + 64 <= D_) {
      __builtin_prefetch(A0 + kk + 32, 0, 3);
      __builtin_prefetch(A1 + kk + 32, 0, 3);
    }
    v16bf b0 = load_ab16x32(Bs + kk, D_);           // ds_load_b128 x2
    v16bf b1 = load_ab16x32(Bs + 16 * D_ + kk, D_);
    acc00 = wmma_bf16(a0, b0, acc00);
    acc01 = wmma_bf16(a0, b1, acc01);
    acc10 = wmma_bf16(a1, b0, acc10);
    acc11 = wmma_bf16(a1, b1, acc11);
    a0 = na0; a1 = na1;
  }

  const int nl = lane & 15, half = lane >> 4;
  #pragma unroll
  for (int j = 0; j < 2; ++j) {
    const int n = n0 + j * 16 + nl;
    const float bvv = bias ? bias[n] : 0.0f;
    const v8f accj0 = j ? acc01 : acc00;
    const v8f accj1 = j ? acc11 : acc10;
    #pragma unroll
    for (int i = 0; i < 8; ++i) {
      int ma = m0 + i + 8 * half;        // first m-tile
      int mb = ma + 16;                  // second m-tile
      float va = accj0[i] + bvv;
      float vb = accj1[i] + bvv;
      if (mode == 0) {
        Cb[(size_t)ma * ldc + n] = (bf16)va;
        Cb[(size_t)mb * ldc + n] = (bf16)vb;
      } else if (mode == 1) {
        int ba = ma >> 10, ka = ma & (KLEN - 1);
        int bb = mb >> 10, kb = mb & (KLEN - 1);
        Cb[((size_t)ba * D_ + n) * KLEN + ka] = (bf16)va;
        Cb[((size_t)bb * D_ + n) * KLEN + kb] = (bf16)vb;
      } else {
        Cf[(size_t)ma * ldc + n] = va;
        Cf[(size_t)mb * ldc + n] = vb;
      }
    }
  }
}

// ---------------- batched head-energy GEMM ------------------------------
// grid (KLEN/32, 1, B_*H), block (32,8); wave ty owns q-tile ty, 1x2 n-tiles.
// 32x128 K-panel async-staged to LDS (8 KB), shared by all 8 waves.
// mode 0: monotonic -> out = sigmoid(e/scale + r) ; mode 1: chunk -> e/scale
__global__ void energy_wmma_kernel(const bf16* __restrict__ Qh,
                                   const bf16* __restrict__ Kh,
                                   const int* __restrict__ mask,
                                   const float* __restrict__ rptr,
                                   float* __restrict__ Out, int mode) {
  __shared__ __align__(16) bf16 Bs[32 * DK_];         // 8 KB
  const int lane = threadIdx.x;
  const int z = blockIdx.z;                 // b*H + h
  const int b = z / HM_, h = z % HM_;
  const int q0 = threadIdx.y * 16;
  const int n0 = blockIdx.x * 32;
  const int t  = threadIdx.y * 32 + lane;

  // ---- async-stage K panel: 32 rows x 256B, 512 x 16B chunks ----
  {
    const char* Bg = (const char*)(Kh + ((size_t)(b * KLEN) + n0) * D_ + h * DK_);
    const unsigned lds_base = (unsigned)(size_t)(const void*)&Bs[0];
    #pragma unroll
    for (int it = 0; it < 2; ++it) {
      int idx = it * 256 + t;
      int rn = idx >> 4;              // row in panel (16 chunks per row)
      int cb = idx & 15;
      async_copy_b128(lds_base + (unsigned)(rn * (DK_ * 2) + cb * 16),
                      Bg + (size_t)rn * D_ * 2 + cb * 16);
    }
  }
  wait_async0();
  __syncthreads();

  const bf16* Ab = Qh + ((size_t)(b * QLEN) + q0) * D_ + h * DK_;

  v8f acc0 = {}, acc1 = {};
  v16bf a = load_ab16x32(Ab, D_);
  #pragma unroll
  for (int kk = 0; kk < DK_; kk += 32) {
    const int kn = (kk + 32 < DK_) ? kk + 32 : kk;
    v16bf na = load_ab16x32(Ab + kn, D_);
    v16bf b0 = load_ab16x32(Bs + kk, DK_);
    v16bf b1 = load_ab16x32(Bs + 16 * DK_ + kk, DK_);
    acc0 = wmma_bf16(a, b0, acc0);
    acc1 = wmma_bf16(a, b1, acc1);
    a = na;
  }

  const float inv_scale = 0.044194173824159216f;  // 1/sqrt(512)
  const float radd = (mode == 0) ? rptr[0] : 0.0f;
  const int nl = lane & 15, half = lane >> 4;
  #pragma unroll
  for (int j = 0; j < 2; ++j) {
    const int k = n0 + j * 16 + nl;
    const v8f accj = j ? acc1 : acc0;
    #pragma unroll
    for (int i = 0; i < 8; ++i) {
      int q = q0 + i + 8 * half;
      float e = accj[i] * inv_scale + radd;
      if (mask[((size_t)b * QLEN + q) * KLEN + k] <= 0) e = NEG_INF_;
      float o = (mode == 0) ? (1.0f / (1.0f + expf(-e))) : e;
      Out[((size_t)z * QLEN + q) * KLEN + k] = o;
    }
  }
}

// ---------------- block-wide inclusive scan (1024 thr = 32 wave32) ------
__device__ inline float block_inclusive_scan(float v, float* lds) {
  const int lane = threadIdx.x & 31;
  const int wid  = threadIdx.x >> 5;
  #pragma unroll
  for (int off = 1; off < 32; off <<= 1) {
    float n = __shfl_up(v, off, 32);
    if (lane >= off) v += n;
  }
  __syncthreads();                // protect lds from previous call's readers
  if (lane == 31) lds[wid] = v;
  __syncthreads();
  if (wid == 0) {
    float w = lds[lane];
    #pragma unroll
    for (int off = 1; off < 32; off <<= 1) {
      float n = __shfl_up(w, off, 32);
      if (lane >= off) w += n;
    }
    lds[lane] = w;
  }
  __syncthreads();
  if (wid > 0) v += lds[wid - 1];
  return v;
}

// ---------------- monotonic alpha: sequential over Q, scans over K ------
// grid = B_*HM_ blocks, 1024 threads (one per k).
__global__ void alpha_kernel(const float* __restrict__ P,
                             float* __restrict__ ALPHA) {
  __shared__ float lds[32];
  const int z = blockIdx.x;
  const int k = threadIdx.x;
  const float* Pz = P     + (size_t)z * QLEN * KLEN;
  float*       Az = ALPHA + (size_t)z * QLEN * KLEN;

  float aw_prev = (k == 0) ? 1.0f : 0.0f;
  for (int q = 0; q < QLEN; ++q) {
    float p = Pz[(size_t)q * KLEN + k];
    float l = logf(fminf(fmaxf(1.0f - p, EPS_), 1.0f));
    float inc = block_inclusive_scan(l, lds);
    float cp  = expf(inc - l);                      // exclusive cumsum
    float t   = aw_prev / fmaxf(cp, EPS_);
    float s   = block_inclusive_scan(t, lds);       // inclusive cumsum
    float aw  = p * cp * s;
    Az[(size_t)q * KLEN + k] = aw;
    aw_prev = aw;
  }
}

// ---------------- row max + clamped exp (in place on e_chunk) -----------
// grid = B_*HC_*QLEN rows, 1024 threads.
__global__ void sexp_kernel(float* __restrict__ E) {
  __shared__ float red[32];
  const int k = threadIdx.x;
  const int lane = k & 31, wid = k >> 5;
  float* Er = E + (size_t)blockIdx.x * KLEN;
  float e = Er[k];
  float m = e;
  #pragma unroll
  for (int off = 16; off > 0; off >>= 1) m = fmaxf(m, __shfl_xor(m, off, 32));
  if (lane == 0) red[wid] = m;
  __syncthreads();
  if (wid == 0) {
    float t = red[lane];
    #pragma unroll
    for (int off = 16; off > 0; off >>= 1) t = fmaxf(t, __shfl_xor(t, off, 32));
    red[lane] = t;
  }
  __syncthreads();
  m = red[0];
  Er[k] = fmaxf(expf(e - m), 1e-5f);
}

// ---------------- beta = sexp * movsum_fwd(alpha / movsum_back(sexp)) ---
// grid = B_*HC_*QLEN rows, 1024 threads; loops hm internally. bf16 out.
__global__ void beta_kernel(const float* __restrict__ SEXP,
                            const float* __restrict__ ALPHA,
                            bf16* __restrict__ BETA) {
  __shared__ float ls[KLEN];
  __shared__ float lr[KLEN];
  const int row = blockIdx.x;                 // (b*HC + hc)*QLEN + q
  const int q   = row % QLEN;
  const int bhc = row / QLEN;
  const int hc  = bhc % HC_;
  const int b   = bhc / HC_;
  const int k   = threadIdx.x;

  float s = SEXP[(size_t)row * KLEN + k];
  ls[k] = s;
  __syncthreads();
  float denom = ls[k];                        // back-window CHUNK-1 = 3
  if (k >= 1) denom += ls[k - 1];
  if (k >= 2) denom += ls[k - 2];
  if (k >= 3) denom += ls[k - 3];

  for (int hm = 0; hm < HM_; ++hm) {
    float a = ALPHA[(((size_t)b * HM_ + hm) * QLEN + q) * KLEN + k];
    lr[k] = a / denom;
    __syncthreads();
    float acc = lr[k];                        // forward window 3
    if (k + 1 < KLEN) acc += lr[k + 1];
    if (k + 2 < KLEN) acc += lr[k + 2];
    if (k + 3 < KLEN) acc += lr[k + 3];
    float beta = s * acc;
    BETA[(((size_t)b * HTOT + hm * HC_ + hc) * QLEN + q) * KLEN + k] = (bf16)beta;
    __syncthreads();
  }
}

// ---------------- context: cv[b,q,h,:] = beta[b,h,q,:] @ vt[b,h,:,:]^T --
// grid (1, 1, B_*HTOT=128), block (32,8); wave ty owns q-tile ty; both
// 16-wide n-tiles (VD=32) kept in registers; pipelined over K=1024.
__global__ void context_wmma_kernel(const bf16* __restrict__ BETA,
                                    const bf16* __restrict__ VT,
                                    bf16* __restrict__ CVb) {
  const int lane = threadIdx.x;
  const int z = blockIdx.z;               // b*16 + h
  const int b = z >> 4, h = z & 15;
  const int q0 = threadIdx.y * 16;

  const bf16* Ab = BETA + ((size_t)z * QLEN + q0) * KLEN;
  const bf16* B0 = VT   + ((size_t)b * D_ + h * VD) * KLEN;
  const bf16* B1 = B0 + (size_t)16 * KLEN;

  v8f acc0 = {}, acc1 = {};
  v16bf a  = load_ab16x32(Ab, KLEN);
  v16bf b0 = load_ab16x32(B0, KLEN);
  v16bf b1 = load_ab16x32(B1, KLEN);
  for (int kk = 32; kk < KLEN; kk += 32) {
    v16bf na  = load_ab16x32(Ab + kk, KLEN);
    v16bf nb0 = load_ab16x32(B0 + kk, KLEN);
    v16bf nb1 = load_ab16x32(B1 + kk, KLEN);
    if (kk + 64 <= KLEN) {
      __builtin_prefetch(Ab + kk + 32, 0, 3);
      __builtin_prefetch(B0 + kk + 32, 0, 3);
    }
    acc0 = wmma_bf16(a, b0, acc0);
    acc1 = wmma_bf16(a, b1, acc1);
    a = na; b0 = nb0; b1 = nb1;
  }
  acc0 = wmma_bf16(a, b0, acc0);
  acc1 = wmma_bf16(a, b1, acc1);

  const int nl = lane & 15, half = lane >> 4;
  #pragma unroll
  for (int j = 0; j < 2; ++j) {
    const v8f accj = j ? acc1 : acc0;
    #pragma unroll
    for (int i = 0; i < 8; ++i) {
      int q = q0 + i + 8 * half;
      CVb[((size_t)b * QLEN + q) * D_ + h * VD + j * 16 + nl] = (bf16)accj[i];
    }
  }
}

// ======================================================================
extern "C" void kernel_launch(void* const* d_in, const int* in_sizes, int n_in,
                              void* d_out, int out_size, void* d_ws, size_t ws_size,
                              hipStream_t stream) {
  const float* key_x   = (const float*)d_in[0];
  const float* query_x = (const float*)d_in[1];
  const int*   mask    = (const int*)  d_in[2];
  const float* wk_m = (const float*)d_in[3];  const float* bk_m = (const float*)d_in[4];
  const float* wq_m = (const float*)d_in[5];  const float* bq_m = (const float*)d_in[6];
  const float* rptr = (const float*)d_in[7];
  const float* wk_c = (const float*)d_in[8];  const float* bk_c = (const float*)d_in[9];
  const float* wq_c = (const float*)d_in[10]; const float* bq_c = (const float*)d_in[11];
  const float* wv   = (const float*)d_in[12]; const float* bv   = (const float*)d_in[13];
  const float* wo   = (const float*)d_in[14]; const float* bo   = (const float*)d_in[15];
  float* out = (float*)d_out;

  const int BK = B_ * KLEN;   // 8192
  const int BQ = B_ * QLEN;   // 1024

  // bump allocator on workspace (256B aligned slabs)
  char* wp = (char*)d_ws;
  auto alloc = [&](size_t bytes) -> char* {
    char* r = wp;
    wp += (bytes + 255) & ~(size_t)255;
    return r;
  };
  bf16* keyb  = (bf16*)alloc((size_t)BK * D_ * sizeof(bf16));
  bf16* qryb  = (bf16*)alloc((size_t)BQ * D_ * sizeof(bf16));
  bf16* wt_km = (bf16*)alloc((size_t)D_ * D_ * sizeof(bf16));
  bf16* wt_qm = (bf16*)alloc((size_t)D_ * D_ * sizeof(bf16));
  bf16* wt_kc = (bf16*)alloc((size_t)D_ * D_ * sizeof(bf16));
  bf16* wt_qc = (bf16*)alloc((size_t)D_ * D_ * sizeof(bf16));
  bf16* wt_v  = (bf16*)alloc((size_t)D_ * D_ * sizeof(bf16));
  bf16* wt_o  = (bf16*)alloc((size_t)D_ * D_ * sizeof(bf16));
  bf16* km_b  = (bf16*)alloc((size_t)BK * D_ * sizeof(bf16));
  bf16* kc_b  = (bf16*)alloc((size_t)BK * D_ * sizeof(bf16));
  bf16* qm_b  = (bf16*)alloc((size_t)BQ * D_ * sizeof(bf16));
  bf16* qc_b  = (bf16*)alloc((size_t)BQ * D_ * sizeof(bf16));
  bf16* vt_b  = (bf16*)alloc((size_t)B_ * D_ * KLEN * sizeof(bf16));
  float* P    = (float*)alloc((size_t)B_ * HM_ * QLEN * KLEN * sizeof(float));
  float* ALPH = (float*)alloc((size_t)B_ * HM_ * QLEN * KLEN * sizeof(float));
  float* ECH  = (float*)alloc((size_t)B_ * HC_ * QLEN * KLEN * sizeof(float));
  bf16* BETA  = (bf16*)alloc((size_t)B_ * HTOT * QLEN * KLEN * sizeof(bf16));
  bf16* CVb   = (bf16*)alloc((size_t)BQ * D_ * sizeof(bf16));
  (void)ws_size; (void)n_in; (void)in_sizes; (void)out_size;

  const dim3 gblk(32, 8);

  // ---- downconvert activations & weights ----
  pack_bf16_kernel<<<4096, 256, 0, stream>>>(key_x,   keyb, BK * D_);
  pack_bf16_kernel<<<1024, 256, 0, stream>>>(query_x, qryb, BQ * D_);
  packT_bf16_kernel<<<1024, 256, 0, stream>>>(wk_m, wt_km, D_, D_);
  packT_bf16_kernel<<<1024, 256, 0, stream>>>(wq_m, wt_qm, D_, D_);
  packT_bf16_kernel<<<1024, 256, 0, stream>>>(wk_c, wt_kc, D_, D_);
  packT_bf16_kernel<<<1024, 256, 0, stream>>>(wq_c, wt_qc, D_, D_);
  packT_bf16_kernel<<<1024, 256, 0, stream>>>(wv,   wt_v,  D_, D_);
  packT_bf16_kernel<<<1024, 256, 0, stream>>>(wo,   wt_o,  D_, D_);

  // ---- projections (WMMA bf16, 32x32 per wave, B panel via async LDS) ----
  gemm_wmma_kernel<<<dim3(D_/32, BK/256), gblk, 0, stream>>>(
      keyb, D_, wt_km, D_, bk_m, nullptr, km_b, D_, BK, 0);
  gemm_wmma_kernel<<<dim3(D_/32, BQ/256), gblk, 0, stream>>>(
      qryb, D_, wt_qm, D_, bq_m, nullptr, qm_b, D_, BQ, 0);
  gemm_wmma_kernel<<<dim3(D_/32, BK/256), gblk, 0, stream>>>(
      keyb, D_, wt_kc, D_, bk_c, nullptr, kc_b, D_, BK, 0);
  gemm_wmma_kernel<<<dim3(D_/32, BQ/256), gblk, 0, stream>>>(
      qryb, D_, wt_qc, D_, bq_c, nullptr, qc_b, D_, BQ, 0);
  gemm_wmma_kernel<<<dim3(D_/32, BK/256), gblk, 0, stream>>>(
      keyb, D_, wt_v,  D_, bv,   nullptr, vt_b, 0,  BK, 1);   // transposed store

  // ---- energies (batched per (b,h), WMMA bf16, K panel via async LDS) ----
  energy_wmma_kernel<<<dim3(KLEN/32, 1, B_*HM_), gblk, 0, stream>>>(
      qm_b, km_b, mask, rptr, P,   0);   // p_choose = sigmoid(e/scale + r)
  energy_wmma_kernel<<<dim3(KLEN/32, 1, B_*HC_), gblk, 0, stream>>>(
      qc_b, kc_b, mask, rptr, ECH, 1);   // raw chunk energy

  // ---- monotonic alpha (sequential over Q; block scans over K) ----
  alpha_kernel<<<B_*HM_, KLEN, 0, stream>>>(P, ALPH);

  // ---- chunkwise softmax pieces ----
  sexp_kernel<<<B_*HC_*QLEN, KLEN, 0, stream>>>(ECH);         // in-place sexp
  beta_kernel<<<B_*HC_*QLEN, KLEN, 0, stream>>>(ECH, ALPH, BETA);

  // ---- context einsum (128 batched WMMA GEMMs 128x1024x32) ----
  context_wmma_kernel<<<dim3(1, 1, B_*HTOT), gblk, 0, stream>>>(BETA, vt_b, CVb);

  // ---- output projection -> f32 d_out ----
  gemm_wmma_kernel<<<dim3(D_/32, BQ/256), gblk, 0, stream>>>(
      CVb, D_, wt_o, D_, bo, out, nullptr, D_, BQ, 2);
}
